// AugmentedLstm_49752901157545
// MI455X (gfx1250) — matrix operand order repaired and checked by reading, not compiled, not run on hardware
//
#include <hip/hip_runtime.h>
#include <hip/hip_bf16.h>

#define Bn     64
#define Tn     512
#define Dn     512
#define Hn     512
#define SIXH   3072
#define FIVEH  2560
#define BT     32768      // Bn*Tn

typedef __attribute__((ext_vector_type(16))) __bf16 v16bf;
typedef __attribute__((ext_vector_type(8)))  float  v8f;

// ---------------- workspace layout (bytes) ----------------
#define OFF_XB    ((size_t)0)                               // x as bf16: 32 MB
#define OFF_WIN   (OFF_XB   + (size_t)BT * Dn * 2)          // w_in bf16: 3 MB
#define OFF_WST   (OFF_WIN  + (size_t)SIXH * Dn * 2)        // w_state bf16: 2.5 MB
#define OFF_PROJ  (OFF_WST  + (size_t)FIVEH * Dn * 2)       // proj bf16: 192 MB
#define OFF_H     (OFF_PROJ + (size_t)BT * SIXH * 2)        // h double buffers: 128 KB
#define OFF_CTR   (OFF_H    + (size_t)4 * 2 * 16 * Hn * 2)  // 4 group counters (padded)

// ---------------- WMMA helpers ----------------
__device__ __forceinline__ v16bf pack16(uint4 lo, uint4 hi) {
    union { uint4 u[2]; v16bf v; } un;
    un.u[0] = lo; un.u[1] = hi;
    return un.v;
}
// A-matrix 16x32 bf16 tile: lanes 0-15 -> K {0..7, 16..23}; lanes 16-31 -> K {8..15, 24..31}
__device__ __forceinline__ v16bf load_a16(const __bf16* p, int ld, int row, int k0, int half) {
    const __bf16* r = p + (size_t)row * ld + k0 + half * 8;
    return pack16(*(const uint4*)r, *(const uint4*)(r + 16));
}
// B-matrix 32x16 bf16 tile: lanes 0-15 -> K {0..7, 16..23}; lanes 16-31 -> K {8..15, 24..31}
__device__ __forceinline__ v16bf load_b16(const __bf16* p, int ld, int col, int k0, int half) {
    const __bf16* r = p + (size_t)col * ld + k0 + half * 8;
    return pack16(*(const uint4*)r, *(const uint4*)(r + 16));
}
__device__ __forceinline__ v8f wmma_bf16(v16bf a, v16bf b, v8f c) {
    return __builtin_amdgcn_wmma_f32_16x16x32_bf16(false, a, false, b, (short)0, c, false, false);
}
__device__ __forceinline__ float sigm(float x) { return 1.0f / (1.0f + __expf(-x)); }

// ---------------- kernel 0: convert inputs to bf16, zero h buffers + counters ----------------
__global__ void prep(const float* __restrict__ x, const float* __restrict__ w_in,
                     const float* __restrict__ w_state,
                     __bf16* __restrict__ xb, __bf16* __restrict__ winb,
                     __bf16* __restrict__ wstb, __bf16* __restrict__ hbuf,
                     unsigned* __restrict__ ctr)
{
    size_t i = (size_t)blockIdx.x * blockDim.x + threadIdx.x;
    if (i < (size_t)BT * Dn)        xb[i]   = (__bf16)x[i];
    if (i < (size_t)SIXH * Dn)      winb[i] = (__bf16)w_in[i];
    if (i < (size_t)FIVEH * Dn)     wstb[i] = (__bf16)w_state[i];
    if (i < (size_t)4 * 2 * 16 * Hn) hbuf[i] = (__bf16)0.0f;
    if (i < 64)                     ctr[i]  = 0u;
}

// ---------------- kernel 1: proj = x @ w_in^T + b_in  (bf16 WMMA, f32 accum) ----------------
// one wave computes a 32(M) x 64(N) block: 2 M-tiles x 4 N-tiles, K=512
__global__ void __launch_bounds__(256) proj_gemm(const __bf16* __restrict__ xb,
                                                 const __bf16* __restrict__ winb,
                                                 const float*  __restrict__ b_in,
                                                 __bf16* __restrict__ proj)
{
    const int lane = threadIdx.x & 31;
    const int wid  = blockIdx.x * (blockDim.x >> 5) + (threadIdx.x >> 5);
    const int nstrip = wid % 48;        // 48 strips of 64 cols = 3072
    const int mpair  = wid / 48;        // 1024 pairs of 16 rows = 32768
    const int m0 = mpair * 32;
    const int n0 = nstrip * 64;
    const int half = lane >> 4, idx = lane & 15;

    v8f acc[2][4] = {};
    for (int k0 = 0; k0 < Dn; k0 += 32) {
        v16bf a0 = load_a16(xb, Dn, m0 + idx,      k0, half);
        v16bf a1 = load_a16(xb, Dn, m0 + 16 + idx, k0, half);
#pragma unroll
        for (int j = 0; j < 4; ++j) {
            v16bf b = load_b16(winb, Dn, n0 + j * 16 + idx, k0, half);
            acc[0][j] = wmma_bf16(a0, b, acc[0][j]);
            acc[1][j] = wmma_bf16(a1, b, acc[1][j]);
        }
    }
#pragma unroll
    for (int s = 0; s < 2; ++s) {
#pragma unroll
        for (int j = 0; j < 4; ++j) {
            const int col = n0 + j * 16 + idx;
            const float bia = b_in[col];
#pragma unroll
            for (int r = 0; r < 8; ++r) {
                const int row = m0 + s * 16 + r + half * 8;   // C/D layout: vgpr r -> M=r (+8 hi half)
                proj[(size_t)row * SIXH + col] = (__bf16)(acc[s][j][r] + bia);
            }
        }
    }
}

// ---------------- kernel 2: persistent recurrence ----------------
// grid = 64 WGs: 4 batch-groups (M=16 rows) x 16 column-slices (32 h-cols each).
// w_state slice (160 rows x 512, padded) resident in LDS; h double-buffered in L2 (bf16);
// groups synchronize per step with a release/acquire counter.
#define HPAD 520
#define SMEM_BYTES (160 * HPAD * 2 + 10 * 256 * 4)

__global__ void lstm_rec(const __bf16* __restrict__ wstb,
                         const __bf16* __restrict__ proj,
                         const float*  __restrict__ b_state,
                         const int*    __restrict__ lengths,
                         float* __restrict__ out,
                         __bf16* __restrict__ hbuf,
                         unsigned* __restrict__ ctr)
{
    extern __shared__ char smem[];
    __bf16* Wl  = (__bf16*)smem;                         // [160][HPAD] bf16
    float*  psb = (float*)(smem + 160 * HPAD * 2);       // [10 tiles][16*16] f32

    const int g    = blockIdx.x >> 4;    // batch group 0..3
    const int sub  = blockIdx.x & 15;    // column slice 0..15
    const int tid  = threadIdx.x;
    const int wave = tid >> 5;           // 10 waves: wave = chunk*2 + colhalf
    const int lane = tid & 31;
    const int half = lane >> 4;
    const int idx  = lane & 15;

    // stage w_state slice: local row lr -> global row chunk*512 + sub*32 + j
    for (int e = tid; e < 160 * Dn; e += blockDim.x) {
        int lr = e >> 9, k = e & (Dn - 1);
        int chunk = lr >> 5, j = lr & 31;
        Wl[lr * HPAD + k] = wstb[(size_t)(chunk * Hn + sub * 32 + j) * Dn + k];
    }
    __syncthreads();

    const int lrow = wave * 16 + idx;    // this wave's B column in LDS row units

    // gate-wave persistent state (waves 0,1 own the 2 sub-tiles of 16 h-cols)
    float cs[8];
    int   lenr[8];
    float bs[5];
    if (wave < 2) {
#pragma unroll
        for (int r = 0; r < 8; ++r) {
            cs[r]   = 0.0f;
            lenr[r] = lengths[g * 16 + r + half * 8];
        }
#pragma unroll
        for (int c5 = 0; c5 < 5; ++c5)
            bs[c5] = b_state[c5 * Hn + sub * 32 + wave * 16 + idx];
    }

    unsigned* cptr = ctr + g * 16;       // 64B-padded per-group counter

    for (int t = 0; t < Tn; ++t) {
        if (t > 0) {
            if (tid == 0) {
                while (__hip_atomic_load(cptr, __ATOMIC_ACQUIRE, __HIP_MEMORY_SCOPE_AGENT)
                       < (unsigned)(16 * t)) {
                    __builtin_amdgcn_s_sleep(1);
                }
            }
            __syncthreads();
        }
        const __bf16* hin = hbuf + (size_t)(g * 2 + (t & 1)) * 16 * Hn;

        // ps tile for this wave: A = h[16x512] (bf16 from L2), B = w_state slice (LDS)
        v8f acc = {};
#pragma unroll 4
        for (int k0 = 0; k0 < Dn; k0 += 32) {
            v16bf a = load_a16(hin, Hn, idx, k0, half);
            v16bf b = load_b16(Wl, HPAD, lrow, k0, half);
            acc = wmma_bf16(a, b, acc);
        }
#pragma unroll
        for (int r = 0; r < 8; ++r)
            psb[wave * 256 + (r + half * 8) * 16 + idx] = acc[r];
        __syncthreads();

        if (wave < 2) {
            __bf16* hout = hbuf + (size_t)(g * 2 + ((t + 1) & 1)) * 16 * Hn;
            const int pc = sub * 32 + wave * 16 + idx;        // global h column
#pragma unroll
            for (int r = 0; r < 8; ++r) {
                const int m = r + half * 8;
                const size_t prow = ((size_t)(g * 16 + m) * Tn + t) * SIXH;
                const float pi0 = (float)proj[prow + 0 * Hn + pc];
                const float pi1 = (float)proj[prow + 1 * Hn + pc];
                const float pi2 = (float)proj[prow + 2 * Hn + pc];
                const float pi3 = (float)proj[prow + 3 * Hn + pc];
                const float pi4 = (float)proj[prow + 4 * Hn + pc];
                const float pi5 = (float)proj[prow + 5 * Hn + pc];
                const int mi = m * 16 + idx;
                const float ps0 = psb[(0 * 2 + wave) * 256 + mi] + bs[0];
                const float ps1 = psb[(1 * 2 + wave) * 256 + mi] + bs[1];
                const float ps2 = psb[(2 * 2 + wave) * 256 + mi] + bs[2];
                const float ps3 = psb[(3 * 2 + wave) * 256 + mi] + bs[3];
                const float ps4 = psb[(4 * 2 + wave) * 256 + mi] + bs[4];
                const float ig = sigm(pi0 + ps0);
                const float fg = sigm(pi1 + ps1);
                const float gg = tanhf(pi2 + ps2);
                const float og = sigm(pi3 + ps3);
                const float hw = sigm(pi4 + ps4);
                float cn = ig * gg + fg * cs[r];
                float ov = og * tanhf(cn);
                ov = hw * ov + (1.0f - hw) * pi5;
                const bool valid = t < lenr[r];
                ov = valid ? ov : 0.0f;
                cn = valid ? cn : 0.0f;
                cs[r] = cn;
                out[((size_t)(g * 16 + m) * Tn + t) * Hn + pc] = ov;
                hout[m * Hn + pc] = (__bf16)ov;
            }
            __threadfence();   // make h(t) visible at agent scope before signaling
        }
        __syncthreads();
        if (tid == 0)
            __hip_atomic_fetch_add(cptr, 1u, __ATOMIC_RELEASE, __HIP_MEMORY_SCOPE_AGENT);
    }
}

// ---------------- host launcher ----------------
extern "C" void kernel_launch(void* const* d_in, const int* in_sizes, int n_in,
                              void* d_out, int out_size, void* d_ws, size_t ws_size,
                              hipStream_t stream)
{
    const float* x       = (const float*)d_in[0];
    const int*   lengths = (const int*)d_in[1];
    const float* w_in    = (const float*)d_in[2];
    const float* b_in    = (const float*)d_in[3];
    const float* w_state = (const float*)d_in[4];
    const float* b_state = (const float*)d_in[5];
    float* out = (float*)d_out;
    char*  ws  = (char*)d_ws;

    __bf16*   xb   = (__bf16*)(ws + OFF_XB);
    __bf16*   winb = (__bf16*)(ws + OFF_WIN);
    __bf16*   wstb = (__bf16*)(ws + OFF_WST);
    __bf16*   proj = (__bf16*)(ws + OFF_PROJ);
    __bf16*   hbuf = (__bf16*)(ws + OFF_H);
    unsigned* ctr  = (unsigned*)(ws + OFF_CTR);

    (void)hipFuncSetAttribute((const void*)lstm_rec,
                              hipFuncAttributeMaxDynamicSharedMemorySize, SMEM_BYTES);

    prep<<<65536, 256, 0, stream>>>(x, w_in, w_state, xb, winb, wstb, hbuf, ctr);
    proj_gemm<<<6144, 256, 0, stream>>>(xb, winb, b_in, proj);
    lstm_rec<<<64, 320, SMEM_BYTES, stream>>>(wstb, proj, b_state, lengths, out, hbuf, ctr);
}
